// EmbeddingHAMLET_9852654977596
// MI455X (gfx1250) — matrix operand order repaired and checked by reading, not compiled, not run on hardware
//
#include <hip/hip_runtime.h>
#include <math.h>

// ---------------- types ----------------
typedef __bf16 v16bf __attribute__((ext_vector_type(16)));
typedef __bf16 v8bf  __attribute__((ext_vector_type(8)));
typedef float  v8f   __attribute__((ext_vector_type(8)));
typedef int    v4i   __attribute__((vector_size(16)));

#define GLOBAL_AS __attribute__((address_space(1)))
#define LDS_AS    __attribute__((address_space(3)))

#if __has_builtin(__builtin_amdgcn_global_load_async_to_lds_b128)
#define HAVE_ASYNC_LDS 1
#else
#define HAVE_ASYNC_LDS 0
#endif

__device__ __forceinline__ unsigned short f2bf(float f) {
    union { float f; unsigned u; } v; v.f = f;
    unsigned r = v.u + 0x7FFFu + ((v.u >> 16) & 1u);
    return (unsigned short)(r >> 16);
}
__device__ __forceinline__ float bf2f(unsigned short h) {
    union { unsigned u; float f; } v; v.u = ((unsigned)h) << 16; return v.f;
}

// 16-byte global -> LDS copy (async on CDNA5 when available)
__device__ __forceinline__ void copy16_g2l(const unsigned short* g, unsigned short* l) {
#if HAVE_ASYNC_LDS
    __builtin_amdgcn_global_load_async_to_lds_b128(
        (GLOBAL_AS v4i*)g, (LDS_AS v4i*)l, 0, 0);
#else
    *(uint4*)l = *(const uint4*)g;
#endif
}
__device__ __forceinline__ void wait_async_copies() {
#if HAVE_ASYNC_LDS
#if __has_builtin(__builtin_amdgcn_s_wait_asynccnt)
    __builtin_amdgcn_s_wait_asynccnt(0);
#else
    asm volatile("s_wait_asynccnt 0x0" ::: "memory");
#endif
#endif
}

// ---------------- fp32 -> bf16 convert ----------------
__global__ void cvt_kernel(const float* __restrict__ s, unsigned short* __restrict__ d, long long n) {
    long long i = (long long)blockIdx.x * blockDim.x + threadIdx.x;
    if (i < n) d[i] = f2bf(s[i]);
}

// ---------------- general bf16 WMMA GEMM:  OUT[M,N] = A[M,K] @ W[N,K]^T ----------------
// act: 0=none, 1=relu, 2=softplus.  A may be split (A0 has K0 cols, A1 the rest).
// 128x128 block tile, 8 waves (wave32), each wave 32x64 => 2x4 v_wmma_f32_16x16x32_bf16.
// LDS double-buffered; tiles streamed with GLOBAL_LOAD_ASYNC_TO_LDS_B128 (ASYNCcnt).
__global__ void __launch_bounds__(256) gemm_bf16_wmma(
    const unsigned short* __restrict__ A0, int K0,
    const unsigned short* __restrict__ A1,
    const unsigned short* __restrict__ Wt,
    const float* __restrict__ bias,
    const float* __restrict__ resid,
    float* __restrict__ outF,
    unsigned short* __restrict__ outB,
    int Mrows, int N, int K, int act)
{
    __shared__ __align__(16) unsigned short tA[2][128 * 32];
    __shared__ __align__(16) unsigned short tB[2][128 * 32];

    const int t     = threadIdx.x;
    const int wave  = t >> 5;
    const int lane  = t & 31;
    const int half  = lane >> 4;
    const int lr    = lane & 15;
    const int waveM = wave >> 1;     // 0..3
    const int waveN = wave & 1;      // 0..1
    const long long rowBase = (long long)blockIdx.x * 128;
    const int colBase = blockIdx.y * 128;
    const int r0 = t >> 2;           // this thread stages rows r0 and r0+64
    const int kc = (t & 3) * 8;      // 8-element column chunk within the 32-wide K tile

    v8f zero = {0.f,0.f,0.f,0.f,0.f,0.f,0.f,0.f};
    v8f acc[2][4];
    #pragma unroll
    for (int mi = 0; mi < 2; ++mi)
        #pragma unroll
        for (int ni = 0; ni < 4; ++ni) acc[mi][ni] = zero;

    const int kt_end = (K + 31) / 32;

    // Stage one 128x32 A tile into tA[buf]; the fast/slow choice is wave-uniform.
    auto stageA = [&](int buf, int kbase) {
        const unsigned short* src = nullptr; long long kw = 0, kloc = 0;
        if (kbase + 32 <= K0)                          { src = A0; kw = K0;     kloc = kbase; }
        else if (A1 && kbase >= K0 && kbase + 32 <= K) { src = A1; kw = K - K0; kloc = kbase - K0; }
        if (src && ((kw & 7) == 0)) {
            copy16_g2l(src + (size_t)(rowBase + r0)      * kw + kloc + kc, &tA[buf][r0 * 32 + kc]);
            copy16_g2l(src + (size_t)(rowBase + r0 + 64) * kw + kloc + kc, &tA[buf][(r0 + 64) * 32 + kc]);
        } else {
            #pragma unroll
            for (int c = 0; c < 2; ++c) {
                int r = r0 + c * 64;
                union { uint4 v; unsigned short u[8]; } tmp;
                #pragma unroll
                for (int e = 0; e < 8; ++e) {
                    int k = kbase + kc + e; unsigned short val = 0;
                    if (k < K0)           val = A0[(size_t)(rowBase + r) * (size_t)K0 + k];
                    else if (A1 && k < K) val = A1[(size_t)(rowBase + r) * (size_t)(K - K0) + (k - K0)];
                    tmp.u[e] = val;
                }
                *(uint4*)&tA[buf][r * 32 + kc] = tmp.v;
            }
        }
    };
    // Stage one 128x32 B tile (rows of W = output columns).
    auto stageB = [&](int buf, int kbase) {
        if (colBase + 128 <= N && kbase + 32 <= K && ((K & 7) == 0)) {
            copy16_g2l(Wt + (size_t)(colBase + r0)      * (size_t)K + kbase + kc, &tB[buf][r0 * 32 + kc]);
            copy16_g2l(Wt + (size_t)(colBase + r0 + 64) * (size_t)K + kbase + kc, &tB[buf][(r0 + 64) * 32 + kc]);
        } else {
            #pragma unroll
            for (int c = 0; c < 2; ++c) {
                int r = r0 + c * 64;
                int n = colBase + r;
                union { uint4 v; unsigned short u[8]; } tmp;
                #pragma unroll
                for (int e = 0; e < 8; ++e) {
                    int k = kbase + kc + e;
                    tmp.u[e] = (n < N && k < K) ? Wt[(size_t)n * (size_t)K + k] : (unsigned short)0;
                }
                *(uint4*)&tB[buf][r * 32 + kc] = tmp.v;
            }
        }
    };

    stageA(0, 0);
    stageB(0, 0);

    for (int kt = 0; kt < kt_end; ++kt) {
        wait_async_copies();      // my tile writes for buf 'cur' are done
        __syncthreads();          // everyone's are done; prev buf reads finished
        const int cur = kt & 1;
        if (kt + 1 < kt_end) {    // stream next tile into the other buffer
            stageA(cur ^ 1, (kt + 1) * 32);
            stageB(cur ^ 1, (kt + 1) * 32);
        }

        // A fragment: lane(0-15) K {0-7,16-23}; lane(16-31) K {8-15,24-31}
        v16bf afrag[2];
        #pragma unroll
        for (int mi = 0; mi < 2; ++mi) {
            int m = waveM * 32 + mi * 16 + lr;
            v8bf lo = *(const v8bf*)&tA[cur][m * 32 + half * 8];
            v8bf hi = *(const v8bf*)&tA[cur][m * 32 + 16 + half * 8];
            afrag[mi] = __builtin_shufflevector(lo, hi, 0,1,2,3,4,5,6,7,8,9,10,11,12,13,14,15);
        }
        // B fragment: lane(0-15) K {0-7,16-23}; lane(16-31) K {8-15,24-31} (column n = lr)
        v16bf bfrag[4];
        #pragma unroll
        for (int ni = 0; ni < 4; ++ni) {
            int n = waveN * 64 + ni * 16 + lr;
            v8bf lo = *(const v8bf*)&tB[cur][n * 32 + half * 8];
            v8bf hi = *(const v8bf*)&tB[cur][n * 32 + 16 + half * 8];
            bfrag[ni] = __builtin_shufflevector(lo, hi, 0,1,2,3,4,5,6,7,8,9,10,11,12,13,14,15);
        }
        #pragma unroll
        for (int mi = 0; mi < 2; ++mi)
            #pragma unroll
            for (int ni = 0; ni < 4; ++ni)
                acc[mi][ni] = __builtin_amdgcn_wmma_f32_16x16x32_bf16(
                    false, afrag[mi], false, bfrag[ni], (short)0, acc[mi][ni], false, false);
    }

    // epilogue: D element (v, lane) -> row = 8*half + v, col = lr
    #pragma unroll
    for (int mi = 0; mi < 2; ++mi) {
        #pragma unroll
        for (int ni = 0; ni < 4; ++ni) {
            int n = colBase + waveN * 64 + ni * 16 + lr;
            if (n >= N) continue;
            float bv = bias ? bias[n] : 0.f;
            #pragma unroll
            for (int v = 0; v < 8; ++v) {
                long long m = rowBase + waveM * 32 + mi * 16 + half * 8 + v;
                float val = acc[mi][ni][v] + bv;
                if (resid) val += resid[(size_t)m * N + n];
                if (act == 1)      val = val > 0.f ? val : 0.f;
                else if (act == 2) val = (val > 20.f) ? val : log1pf(expf(val));
                if (outF) outF[(size_t)m * N + n] = val;
                if (outB) outB[(size_t)m * N + n] = f2bf(val);
            }
        }
    }
}

// ---------------- reps + z builder ----------------
// reps[i,b,h] = mask[b,i] ? x[i,b,h] : (sum_{j!=i} mf[j]*rec_i[j,b,h]) / max(ratio,1)
__global__ void reps_kernel(const float* __restrict__ x, const unsigned char* __restrict__ mask,
                            const float* __restrict__ muys, float* __restrict__ reps,
                            unsigned short* __restrict__ zbf)
{
    long long idx = (long long)blockIdx.x * blockDim.x + threadIdx.x;  // over 4*8192*1024
    if (idx >= (long long)4 * 8192 * 1024) return;
    int i = (int)(idx >> 23);
    long long rem = idx & ((1ll << 23) - 1);
    int b = (int)(rem >> 10);
    int h = (int)(rem & 1023);
    float mf[4];
    #pragma unroll
    for (int j = 0; j < 4; ++j) mf[j] = mask[b * 4 + j] ? 1.f : 0.f;
    const float* base = muys + (size_t)i * 33554432;   // rec_i block [4,8192,1024]
    float s = 0.f, ratio = 0.f;
    #pragma unroll
    for (int j = 0; j < 4; ++j) {
        s += mf[j] * base[((size_t)j * 8192 + b) * 1024 + h];
        ratio += mf[j];
    }
    s     -= mf[i] * base[((size_t)i * 8192 + b) * 1024 + h];
    ratio -= mf[i];
    float agg = s / fmaxf(ratio, 1.f);
    float val = mask[b * 4 + i] ? x[((size_t)i * 8192 + b) * 1024 + h] : agg;
    reps[idx] = val;
    zbf[((size_t)b * 4 + i) * 1024 + h] = f2bf(val);
}

// ---------------- tiny 4x4 attention (per batch,head block) ----------------
__global__ void __launch_bounds__(128) attn_kernel(const unsigned short* __restrict__ qkv,
                                                   unsigned short* __restrict__ o)
{
    const int b  = blockIdx.x >> 1;
    const int hd = blockIdx.x & 1;
    const int t  = threadIdx.x;
    __shared__ float red[16 * 128];
    __shared__ float scores[16];
    __shared__ float aw[16];

    float part[16];
    #pragma unroll
    for (int p = 0; p < 16; ++p) part[p] = 0.f;

    for (int d = t; d < 512; d += 128) {
        float qv[4], kv[4];
        #pragma unroll
        for (int s = 0; s < 4; ++s) {
            size_t row = ((size_t)b * 4 + s) * 3072;
            qv[s] = bf2f(qkv[row + hd * 512 + d]);
            kv[s] = bf2f(qkv[row + 1024 + hd * 512 + d]);
        }
        #pragma unroll
        for (int s = 0; s < 4; ++s)
            #pragma unroll
            for (int ss = 0; ss < 4; ++ss) part[s * 4 + ss] += qv[s] * kv[ss];
    }
    #pragma unroll
    for (int p = 0; p < 16; ++p) red[p * 128 + t] = part[p];
    __syncthreads();
    if (t < 16) {
        float s = 0.f;
        for (int j = 0; j < 128; ++j) s += red[t * 128 + j];
        scores[t] = s;
    }
    __syncthreads();
    if (t < 4) {
        const float scale = 0.04419417382f;   // 1/sqrt(512)
        float mx = -1e30f;
        #pragma unroll
        for (int ss = 0; ss < 4; ++ss) mx = fmaxf(mx, scores[t * 4 + ss] * scale);
        float sum = 0.f, e[4];
        #pragma unroll
        for (int ss = 0; ss < 4; ++ss) { e[ss] = expf(scores[t * 4 + ss] * scale - mx); sum += e[ss]; }
        #pragma unroll
        for (int ss = 0; ss < 4; ++ss) aw[t * 4 + ss] = e[ss] / sum;
    }
    __syncthreads();
    for (int d = t; d < 512; d += 128) {
        float vv[4];
        #pragma unroll
        for (int ss = 0; ss < 4; ++ss)
            vv[ss] = bf2f(qkv[((size_t)b * 4 + ss) * 3072 + 2048 + hd * 512 + d]);
        #pragma unroll
        for (int s = 0; s < 4; ++s) {
            float ov = 0.f;
            #pragma unroll
            for (int ss = 0; ss < 4; ++ss) ov += aw[s * 4 + ss] * vv[ss];
            o[((size_t)b * 4 + s) * 1024 + hd * 512 + d] = f2bf(ov);
        }
    }
}

// ---------------- sum over S=4 rows ----------------
__global__ void rowsum4_kernel(const unsigned short* __restrict__ ap,
                               float* __restrict__ outF, unsigned short* __restrict__ outB)
{
    long long idx = (long long)blockIdx.x * blockDim.x + threadIdx.x;  // over 8192*1024
    if (idx >= (long long)8192 * 1024) return;
    int b = (int)(idx >> 10);
    int h = (int)(idx & 1023);
    float s = 0.f;
    #pragma unroll
    for (int s4 = 0; s4 < 4; ++s4) s += bf2f(ap[((size_t)b * 4 + s4) * 1024 + h]);
    outF[idx] = s;
    outB[idx] = f2bf(s);
}

// ---------------- BatchNorm (training stats, biased var) + ReLU, one block per column ----------------
__global__ void __launch_bounds__(256) bn_relu_kernel(const float* __restrict__ F,
                                                      const float* __restrict__ g,
                                                      const float* __restrict__ be,
                                                      unsigned short* __restrict__ G,
                                                      int N, int rows)
{
    const int n = blockIdx.x;
    const int t = threadIdx.x;
    __shared__ float s1[256], s2[256], par[2];
    float a = 0.f, b2 = 0.f;
    for (int r = t; r < rows; r += 256) {
        float v = F[(size_t)r * N + n];
        a += v; b2 += v * v;
    }
    s1[t] = a; s2[t] = b2;
    __syncthreads();
    for (int off = 128; off > 0; off >>= 1) {
        if (t < off) { s1[t] += s1[t + off]; s2[t] += s2[t + off]; }
        __syncthreads();
    }
    if (t == 0) {
        float m   = s1[0] / (float)rows;
        float var = s2[0] / (float)rows - m * m;
        float k   = g[n] * rsqrtf(var + 1e-5f);
        par[0] = k; par[1] = be[n] - m * k;
    }
    __syncthreads();
    float sc = par[0], sh = par[1];
    for (int r = t; r < rows; r += 256) {
        float v = F[(size_t)r * N + n] * sc + sh;
        v = v > 0.f ? v : 0.f;
        G[(size_t)r * N + n] = f2bf(v);
    }
}

// ---------------- host launcher ----------------
extern "C" void kernel_launch(void* const* d_in, const int* in_sizes, int n_in,
                              void* d_out, int out_size, void* d_ws, size_t ws_size,
                              hipStream_t stream) {
    (void)in_sizes; (void)n_in; (void)out_size; (void)ws_size;
    const int Hh = 1024, Mm = 4, Bb = 8192;
    const long long MBH = (long long)Mm * Bb * Hh;        // 33,554,432

    const float* x = (const float*)d_in[0];
    const unsigned char* mask = (const unsigned char*)d_in[1];   // jnp bool -> 1 byte

    // d_out regions (fp32)
    float* outp       = (float*)d_out;
    float* o_result   = outp;                         // [8192,12]
    float* o_fused    = outp + 98304;                 // [8192,1]
    float* o_reps     = outp + 106496;                // [4,8192,1024]
    float* o_muys     = outp + 33660928;              // [4,4,8192,1024]
    float* o_sig      = outp + 167878656;             // [4,4,8192,1024]

    // -------- workspace layout --------
    char* ws = (char*)d_ws;
    unsigned short* wbase = (unsigned short*)ws;      // bf16 weights (~48.5MB), 64MB reserved
    char* SB = ws + (64ull << 20);

    // phase-1 scratch
    unsigned short* x_bf   = (unsigned short*)(SB);                      // [4*8192,1024]
    unsigned short* rec_bf = (unsigned short*)(SB + 67108864ull);        // [32768,1024]
    unsigned short* h_a    = (unsigned short*)(SB + 134217728ull);       // [32768,512]
    unsigned short* h_b    = (unsigned short*)(SB + 167772160ull);       // [32768,512]
    unsigned short* sig_h  = (unsigned short*)(SB + 201326592ull);       // [32768,1024]
    // phase-2 scratch (aliases phase-1)
    unsigned short* z_bf    = (unsigned short*)(SB);                     // [32768,1024]
    unsigned short* qkv_bf  = (unsigned short*)(SB + 67108864ull);       // [32768,3072]
    unsigned short* o_bf    = (unsigned short*)(SB + 268435456ull);      // [32768,1024]
    unsigned short* ap_bf   = (unsigned short*)(SB + 335544320ull);      // [32768,1024]
    float*          out_f   = (float*)(SB + 402653184ull);               // [8192,1024]
    unsigned short* out_bf  = (unsigned short*)(SB + 436207616ull);      // [8192,1024]
    float*          Fbuf    = (float*)(SB + 452984832ull);               // <= [8192,512] f32
    unsigned short* Gbuf    = (unsigned short*)(SB + 469762048ull);      // <= [8192,512] bf16
    unsigned short* G2buf   = (unsigned short*)(SB + 478150656ull);      // <= [8192,256] bf16
    unsigned short* Rbuf    = (unsigned short*)(SB + 486539264ull);      // [8192,12] bf16
    unsigned short* Xb      = (unsigned short*)(SB + 486801408ull);      // [8192,1024] bf16

    auto cvt = [&](const void* src, unsigned short* dst, long long n) {
        int blocks = (int)((n + 255) / 256);
        cvt_kernel<<<blocks, 256, 0, stream>>>((const float*)src, dst, n);
    };
    size_t woff = 0;
    auto wcvt = [&](int idx, long long n) -> unsigned short* {
        unsigned short* p = wbase + woff;
        cvt(d_in[idx], p, n);
        woff = (woff + (size_t)n + 7) & ~(size_t)7;
        return p;
    };
    auto gemm = [&](const unsigned short* A0, int K0, const unsigned short* A1,
                    const unsigned short* Wt, const float* bias, const float* resid,
                    float* outF, unsigned short* outB, int Mr, int N, int K, int act) {
        dim3 g(Mr / 128, (N + 127) / 128);
        gemm_bf16_wmma<<<g, 256, 0, stream>>>(A0, K0, A1, Wt, bias, resid, outF, outB, Mr, N, K, act);
    };

    // -------- weight conversion --------
    cvt(x, x_bf, MBH);
    unsigned short *recW1[4], *recW2[4], *recW3[4], *sigW1[4], *sigW2[4], *sigW3[4];
    for (int i = 0; i < 4; ++i) {
        int rb = 2 + 6 * i;
        recW1[i] = wcvt(rb + 0, 512 * 1024);
        recW2[i] = wcvt(rb + 2, 512 * 512);
        recW3[i] = wcvt(rb + 4, 1024 * 512);
        int sb = 26 + 6 * i;
        sigW1[i] = wcvt(sb + 0, 1024 * 2048);
        sigW2[i] = wcvt(sb + 2, 512 * 1024);
        sigW3[i] = wcvt(sb + 4, 1024 * 512);
    }
    unsigned short* attnInW  = wcvt(50, 3072 * 1024);
    unsigned short* attnOutW = wcvt(52, 1024 * 1024);
    unsigned short* jpW1 = wcvt(54, 512 * 1024);
    unsigned short* jpW2 = wcvt(58, 128 * 512);
    unsigned short* jpW3 = wcvt(62, 128 * 128);
    unsigned short* jpW4 = wcvt(66, 32 * 128);
    unsigned short* jpW5 = wcvt(70, 12 * 32);
    unsigned short* juA1 = wcvt(72, 512 * 1036);
    unsigned short* juA2 = wcvt(74, 256 * 512);
    unsigned short* juA3 = wcvt(76, 1024 * 256);
    unsigned short* juC1 = wcvt(78, 512 * 1024);
    unsigned short* juC2 = wcvt(80, 256 * 512);
    unsigned short* juC3 = wcvt(82, 1 * 256);

    // -------- phase 1: per-modality reconstructors + sigma nets --------
    for (int i = 0; i < 4; ++i) {
        int rb = 2 + 6 * i, sb = 26 + 6 * i;
        const float* rB1 = (const float*)d_in[rb + 1];
        const float* rB2 = (const float*)d_in[rb + 3];
        const float* rB3 = (const float*)d_in[rb + 5];
        const float* sB1 = (const float*)d_in[sb + 1];
        const float* sB2 = (const float*)d_in[sb + 3];
        const float* sB3 = (const float*)d_in[sb + 5];
        gemm(x_bf, 1024, nullptr, recW1[i], rB1, nullptr, nullptr, h_a, 32768, 512, 1024, 1);
        gemm(h_a,  512,  nullptr, recW2[i], rB2, nullptr, nullptr, h_b, 32768, 512, 512, 1);
        gemm(h_b,  512,  nullptr, recW3[i], rB3, nullptr,
             o_muys + (size_t)i * 33554432, rec_bf, 32768, 1024, 512, 0);
        gemm(x_bf, 1024, rec_bf,  sigW1[i], sB1, nullptr, nullptr, sig_h, 32768, 1024, 2048, 1);
        gemm(sig_h,1024, nullptr, sigW2[i], sB2, nullptr, nullptr, h_a, 32768, 512, 1024, 1);
        gemm(h_a,  512,  nullptr, sigW3[i], sB3, nullptr,
             o_sig + (size_t)i * 33554432, nullptr, 32768, 1024, 512, 2);
    }

    // -------- phase 2: reps + attention --------
    {
        int blocks = (int)((MBH + 255) / 256);
        reps_kernel<<<blocks, 256, 0, stream>>>(x, mask, o_muys, o_reps, z_bf);
    }
    gemm(z_bf, 1024, nullptr, attnInW, (const float*)d_in[51], nullptr, nullptr, qkv_bf, 32768, 3072, 1024, 0);
    attn_kernel<<<8192 * 2, 128, 0, stream>>>(qkv_bf, o_bf);
    gemm(o_bf, 1024, nullptr, attnOutW, (const float*)d_in[53], nullptr, nullptr, ap_bf, 32768, 1024, 1024, 0);
    rowsum4_kernel<<<(8192 * 1024) / 256, 256, 0, stream>>>(ap_bf, out_f, out_bf);

    // -------- joint processing (IMU_MLP with BN) --------
    gemm(out_bf, 1024, nullptr, jpW1, (const float*)d_in[55], nullptr, Fbuf, nullptr, 8192, 512, 1024, 0);
    bn_relu_kernel<<<512, 256, 0, stream>>>(Fbuf, (const float*)d_in[56], (const float*)d_in[57], Gbuf, 512, 8192);
    gemm(Gbuf, 512, nullptr, jpW2, (const float*)d_in[59], nullptr, Fbuf, nullptr, 8192, 128, 512, 0);
    bn_relu_kernel<<<128, 256, 0, stream>>>(Fbuf, (const float*)d_in[60], (const float*)d_in[61], G2buf, 128, 8192);
    gemm(G2buf, 128, nullptr, jpW3, (const float*)d_in[63], nullptr, Fbuf, nullptr, 8192, 128, 128, 0);
    bn_relu_kernel<<<128, 256, 0, stream>>>(Fbuf, (const float*)d_in[64], (const float*)d_in[65], Gbuf, 128, 8192);
    gemm(Gbuf, 128, nullptr, jpW4, (const float*)d_in[67], nullptr, Fbuf, nullptr, 8192, 32, 128, 0);
    bn_relu_kernel<<<32, 256, 0, stream>>>(Fbuf, (const float*)d_in[68], (const float*)d_in[69], G2buf, 32, 8192);
    gemm(G2buf, 32, nullptr, jpW5, (const float*)d_in[71], nullptr, o_result, Rbuf, 8192, 12, 32, 0);

    // -------- joint uncertainty --------
    gemm(out_bf, 1024, Rbuf, juA1, (const float*)d_in[73], nullptr, nullptr, Gbuf, 8192, 512, 1036, 1);
    gemm(Gbuf, 512, nullptr, juA2, (const float*)d_in[75], nullptr, nullptr, G2buf, 8192, 256, 512, 1);
    gemm(G2buf, 256, nullptr, juA3, (const float*)d_in[77], out_f, nullptr, Xb, 8192, 1024, 256, 0);
    gemm(Xb, 1024, nullptr, juC1, (const float*)d_in[79], nullptr, nullptr, Gbuf, 8192, 512, 1024, 1);
    gemm(Gbuf, 512, nullptr, juC2, (const float*)d_in[81], nullptr, nullptr, G2buf, 8192, 256, 512, 1);
    gemm(G2buf, 256, nullptr, juC3, (const float*)d_in[83], nullptr, o_fused, nullptr, 8192, 1, 256, 2);
}
